// VQModel_23244363006173
// MI455X (gfx1250) — matrix operand order repaired and checked by reading, not compiled, not run on hardware
//
#include <hip/hip_runtime.h>
#include <hip/hip_bf16.h>

typedef __attribute__((ext_vector_type(16))) __bf16 v16bf;
typedef __attribute__((ext_vector_type(8)))  __bf16 v8bf;
typedef __attribute__((ext_vector_type(8)))  float  v8f;

#define TPB 256

union Pack8 { __bf16 h[8]; uint4 u; };

// LDS byte-offset of a generic pointer to a __shared__ object
__device__ inline unsigned lds_off_of(const void* p) {
  return (unsigned)(unsigned long long)
      (__attribute__((address_space(3))) const char*)p;
}

// ---------------------------------------------------------------- weight repack:
// OIHW f32 [Cout][Cin][9] -> tap-major bf16 [Cout][9][Cin]  (K = tap*Cin + ci)
__global__ void k_cvt_w_tap(const float* __restrict__ src, __bf16* __restrict__ dst,
                            int Cin, int cinLog2, int total) {
  int i = blockIdx.x * blockDim.x + threadIdx.x;
  if (i >= total) return;
  const int K   = 9 << cinLog2;
  const int co  = i / K;
  const int rem = i - co * K;
  const int tap = rem >> cinLog2;
  const int ci  = rem & (Cin - 1);
  dst[i] = (__bf16)src[((size_t)co * Cin + ci) * 9 + tap];
}

// ---------------------------------------------------------------- first conv: Cin=1 -> NHWC out
__global__ void k_conv3x3_c1(const float* __restrict__ in, const float* __restrict__ w,
                             const float* __restrict__ bias, __bf16* __restrict__ out,
                             int total) {
  int i = blockIdx.x * blockDim.x + threadIdx.x;
  if (i >= total) return;
  const int co  = i & 127;
  const int pix = (i >> 7) & 65535;
  const int b   = i >> 23;
  const int y = pix >> 8, x = pix & 255;
  const float* ip = in + (size_t)b * 65536;
  const float* wp = w + co * 9;
  float s = bias[co];
  #pragma unroll
  for (int ky = 0; ky < 3; ++ky) {
    const int yy = y + ky - 1;
    const bool yok = (unsigned)yy < 256u;
    const int yyc = yok ? yy : 0;
    #pragma unroll
    for (int kx = 0; kx < 3; ++kx) {
      const int xx = x + kx - 1;
      const bool ok = yok && ((unsigned)xx < 256u);
      const int xxc = (unsigned)xx < 256u ? xx : 0;
      const float v = ip[(yyc << 8) + xxc];
      s += (ok ? v : 0.f) * wp[ky * 3 + kx];
    }
  }
  out[i] = (__bf16)s;
}

// ---------------------------------------------------------------- conv3x3 implicit GEMM (WMMA bf16)
// NHWC activations, tap-major K: a thread's 8 K values are 8 consecutive channels
// at one pixel => contiguous 16B. Non-fused convs DMA the B tile straight to LDS
// (global_load_async_to_lds_b128); fused convs use 2x b128 loads + v_pk_add_bf16.
// Double-buffered LDS, one barrier + one asynccnt wait per K-step.
__global__ __launch_bounds__(256)
void k_conv3x3_wmma(const __bf16* __restrict__ in,      // NHWC full-res (skip when fuse_up)
                    const __bf16* __restrict__ lowres,  // NHWC half-res when fuse_up
                    const __bf16* __restrict__ wgt,     // [Cout, 9*Cin] bf16, tap-major
                    const float*  __restrict__ bias,    // [Cout] f32
                    __bf16* __restrict__ out,           // NHWC [B,H,W,Cout] bf16
                    int cinLog2, int Cout, int H, int Wlog2,
                    int relu, int fuse_up)
{
  const int W    = 1 << Wlog2;
  const int P    = H << Wlog2;
  const int Ktot = 9 << cinLog2;
  const int cPerTap = 1 << (cinLog2 - 5);

  const int pixBase = blockIdx.x * 64;
  const int coBase  = blockIdx.y * 64;
  const int b       = blockIdx.z;

  const int t    = threadIdx.x;
  const int lane = t & 31;
  const int wid  = t >> 5;
  const int chSub  = (wid >> 1) << 4;   // 0,16,32,48
  const int pixSub = (wid & 1) << 5;    // 0,32

  __shared__ __align__(16) __bf16 As[2][64 * 40];
  __shared__ __align__(16) __bf16 Bs[2][64 * 40];

  // B-tile role: pixel pl, 8 consecutive channels within one tap
  const int pl  = t >> 2;               // 0..63
  const int bkq = (t & 3) << 3;         // 0,8,16,24
  const int gp  = pixBase + pl;
  const int py  = gp >> Wlog2;          // uniform: tile sits in one image row
  const int px  = gp & (W - 1);
  const unsigned b0o = lds_off_of(&Bs[0][pl * 40 + bkq]);
  const unsigned b1o = lds_off_of(&Bs[1][pl * 40 + bkq]);

  // A-tile role: channel row, 8 K values, async-staged to LDS
  const int arow = t >> 2;
  const int akq  = (t & 3) << 3;
  const unsigned a0 = lds_off_of(&As[0][arow * 40 + akq]);
  const unsigned a1 = lds_off_of(&As[1][arow * 40 + akq]);
  const __bf16* wcur = wgt + (size_t)(coBase + arow) * Ktot + akq;

  // tap-state (advances incrementally; geometry recomputed 9x total)
  int tap = 0, cic = 0, okv = 0;
  const __bf16 *sp = nullptr, *lp = nullptr;
  auto tapSetup = [&](int tp) {
    const int ky = tp < 3 ? 0 : (tp < 6 ? 1 : 2);
    const int kx = tp - ky * 3;
    const int y  = py + ky - 1;
    const int x  = px + kx - 1;
    okv = ((unsigned)y < (unsigned)H) & ((unsigned)x < (unsigned)W);
    const int yc = y < 0 ? 0 : (y >= H ? H - 1 : y);
    const int xc = x < 0 ? 0 : (x >= W ? W - 1 : x);
    sp = in + (((((size_t)(b * H + yc)) << Wlog2) + xc) << cinLog2) + bkq;
    lp = lowres + (((((size_t)(b * (H >> 1) + (yc >> 1))) << (Wlog2 - 1))
                    + (xc >> 1)) << cinLog2) + bkq;
  };
  auto stageB = [&](int buf, __bf16* bufB) {
    if (fuse_up) {                       // skip + up2(lowres): 2x b128 + pk_add_bf16
      v8bf pa, pc;
      __builtin_memcpy(&pa, sp, 16);
      __builtin_memcpy(&pc, lp, 16);
      v8bf pr = pa + pc;
      Pack8 po;
      __builtin_memcpy(&po.u, &pr, 16);
      po.u.x = okv ? po.u.x : 0u;
      po.u.y = okv ? po.u.y : 0u;
      po.u.z = okv ? po.u.z : 0u;
      po.u.w = okv ? po.u.w : 0u;
      *(uint4*)&bufB[pl * 40 + bkq] = po.u;
    } else {                             // pure DMA: global -> LDS, no VGPR round trip
      const unsigned bo = buf ? b1o : b0o;
      if (okv) {
        unsigned long long ga = (unsigned long long)(uintptr_t)sp;
        asm volatile("global_load_async_to_lds_b128 %0, %1, off"
                     :: "v"(bo), "v"(ga) : "memory");
      } else {
        const uint4 z = {0u, 0u, 0u, 0u};
        *(uint4*)&bufB[pl * 40 + bkq] = z;
      }
    }
  };
  auto advance = [&]() {
    if (++cic == cPerTap) { cic = 0; tapSetup(++tap); }
    else { sp += 32; lp += 32; }
  };

  v8f acc0 = {0.f, 0.f, 0.f, 0.f, 0.f, 0.f, 0.f, 0.f};
  v8f acc1 = {0.f, 0.f, 0.f, 0.f, 0.f, 0.f, 0.f, 0.f};

  // fragment LDS row offsets (within a buffer)
  const int ar   = chSub + (lane & 15);
  const int kof  = (lane >> 4) << 3;
  const int kofb = (lane >> 4) << 4;    // A and B: contiguous 16 at kofb
  const int br0  = pixSub + (lane & 15);

  auto compute = [&](const __bf16* Ac, const __bf16* Bc) {
    v16bf af, bf0, bf1;
    __builtin_memcpy((char*)&af,      &Ac[ar * 40 + kofb],     32);
    __builtin_memcpy((char*)&bf0, &Bc[br0 * 40 + kofb], 32);
    __builtin_memcpy((char*)&bf1, &Bc[(br0 + 16) * 40 + kofb], 32);
    acc0 = __builtin_amdgcn_wmma_f32_16x16x32_bf16(
        false, af, false, bf0, (short)0, acc0, false, false);
    acc1 = __builtin_amdgcn_wmma_f32_16x16x32_bf16(
        false, af, false, bf1, (short)0, acc1, false, false);
  };

  const int steps = Ktot >> 5;

  // ---- prologue: stage step 0 into buffer 0
  tapSetup(0);
  {
    unsigned long long ga = (unsigned long long)(uintptr_t)wcur;
    asm volatile("global_load_async_to_lds_b128 %0, %1, off"
                 :: "v"(a0), "v"(ga) : "memory");
    wcur += 32;
  }
  stageB(0, Bs[0]);
  advance();

  // ---- steady state: one barrier per step; next step's tiles staged around the WMMAs
  for (int s = 0; s < steps - 1; ++s) {
    const int cur = s & 1;
    asm volatile("s_wait_asynccnt 0x0" ::: "memory");
    __syncthreads();
    {
      unsigned long long ga = (unsigned long long)(uintptr_t)wcur;
      const unsigned an = cur ? a0 : a1;
      asm volatile("global_load_async_to_lds_b128 %0, %1, off"
                   :: "v"(an), "v"(ga) : "memory");
      __builtin_prefetch(wcur + 32, 0, 0);
      wcur += 32;
    }
    compute(As[cur], Bs[cur]);
    stageB(cur ^ 1, Bs[cur ^ 1]);
    advance();
  }
  asm volatile("s_wait_asynccnt 0x0" ::: "memory");
  __syncthreads();
  compute(As[(steps - 1) & 1], Bs[(steps - 1) & 1]);

  // ---- epilogue: NHWC -> 8 contiguous co per store (2x b128 per thread)
  const int co_lo = coBase + chSub + ((lane >> 4) << 3);
  const int pix0  = pixBase + pixSub + (lane & 15);
  Pack8 o0, o1;
  #pragma unroll
  for (int v2 = 0; v2 < 8; ++v2) {
    const float bv = bias[co_lo + v2];
    float r0 = acc0[v2] + bv;
    float r1 = acc1[v2] + bv;
    if (relu) { r0 = fmaxf(r0, 0.f); r1 = fmaxf(r1, 0.f); }
    o0.h[v2] = (__bf16)r0;
    o1.h[v2] = (__bf16)r1;
  }
  *(uint4*)&out[((size_t)b * P + pix0)      * Cout + co_lo] = o0.u;
  *(uint4*)&out[((size_t)b * P + pix0 + 16) * Cout + co_lo] = o1.u;
}

// ---------------------------------------------------------------- 2x2 maxpool stride 2 (NHWC, 8ch/thread)
__global__ void k_maxpool2(const __bf16* __restrict__ in, __bf16* __restrict__ out,
                           int Hlog2, int Wlog2, int Clog2, int total) {
  int i = blockIdx.x * blockDim.x + threadIdx.x;
  if (i >= total) return;
  const int C  = 1 << Clog2;
  const int cg = i & ((1 << (Clog2 - 3)) - 1);
  int r = i >> (Clog2 - 3);
  const int x = r & ((1 << (Wlog2 - 1)) - 1);
  r >>= (Wlog2 - 1);
  const int y = r & ((1 << (Hlog2 - 1)) - 1);
  const int b = r >> (Hlog2 - 1);
  const size_t base =
      ((((size_t)(b << Hlog2) + 2 * y) << Wlog2) + 2 * x) << Clog2;
  const __bf16* p = in + base + (cg << 3);
  Pack8 q00, q01, q10, q11, o;
  q00.u = *(const uint4*)(p);
  q01.u = *(const uint4*)(p + C);
  q10.u = *(const uint4*)(p + ((size_t)C << Wlog2));
  q11.u = *(const uint4*)(p + ((size_t)C << Wlog2) + C);
  #pragma unroll
  for (int j = 0; j < 8; ++j)
    o.h[j] = (__bf16)fmaxf(fmaxf((float)q00.h[j], (float)q01.h[j]),
                           fmaxf((float)q10.h[j], (float)q11.h[j]));
  *(uint4*)&out[(size_t)i << 3] = o.u;
}

// ---------------------------------------------------------------- 1x1 conv (Cout=1), NHWC in, f32 out
__global__ void k_conv1x1_f32out(const __bf16* __restrict__ in,
                                 const float* __restrict__ w,
                                 const float* __restrict__ b,
                                 float* __restrict__ out,
                                 int Cin, int total) {
  int i = blockIdx.x * blockDim.x + threadIdx.x;
  if (i >= total) return;
  const __bf16* ip = in + (size_t)i * Cin;     // 128 contiguous channels
  float s = b[0];
  for (int c = 0; c < Cin; c += 8) {
    Pack8 pv;
    pv.u = *(const uint4*)&ip[c];
    #pragma unroll
    for (int j = 0; j < 8; ++j) s += (float)pv.h[j] * w[c + j];
  }
  out[i] = s;
}

// ---------------------------------------------------------------- fused qc -> VQ -> pqc
__global__ __launch_bounds__(256)
void k_vq(const float* __restrict__ enc_out,
          const float* __restrict__ qc_w, const float* __restrict__ qc_b,
          const float* __restrict__ cb,
          const float* __restrict__ pqc_w, const float* __restrict__ pqc_b,
          float* __restrict__ dec_in, float* __restrict__ loss_accum) {
  __shared__ float cbs[1024 * 4];
  __shared__ float red[256];
  for (int i = threadIdx.x; i < 4096; i += 256) cbs[i] = cb[i];
  __syncthreads();

  const int i = blockIdx.x * 256 + threadIdx.x;   // exactly 4*65536 threads
  const float s = enc_out[i];
  float z0 = qc_w[0] * s + qc_b[0];
  float z1 = qc_w[1] * s + qc_b[1];
  float z2 = qc_w[2] * s + qc_b[2];
  float z3 = qc_w[3] * s + qc_b[3];

  float bestd = 3.4e38f; int bi = 0;
  for (int e = 0; e < 1024; ++e) {
    const float* c = &cbs[e << 2];
    float d0 = z0 - c[0], d1 = z1 - c[1], d2 = z2 - c[2], d3 = z3 - c[3];
    float d = d0 * d0 + d1 * d1 + d2 * d2 + d3 * d3;
    if (d < bestd) { bestd = d; bi = e; }
  }
  const float* q = &cbs[bi << 2];
  float e0 = q[0] - z0, e1 = q[1] - z1, e2 = q[2] - z2, e3 = q[3] - z3;
  float diff = e0 * e0 + e1 * e1 + e2 * e2 + e3 * e3;

  dec_in[i] = pqc_b[0] + pqc_w[0] * q[0] + pqc_w[1] * q[1]
                       + pqc_w[2] * q[2] + pqc_w[3] * q[3];

  red[threadIdx.x] = diff;
  __syncthreads();
  for (int s2 = 128; s2 > 0; s2 >>= 1) {
    if (threadIdx.x < s2) red[threadIdx.x] += red[threadIdx.x + s2];
    __syncthreads();
  }
  if (threadIdx.x == 0) atomicAdd(loss_accum, red[0]);
}

__global__ void k_loss_out(const float* __restrict__ accum, float* __restrict__ out) {
  out[0] = accum[0] * (1.25f / (262144.0f * 4.0f));
}

// ---------------------------------------------------------------- host side
namespace {

struct UnetW {
  const float  *w0f;
  const __bf16 *wd0, *wd1, *wm, *wu0, *wu1;
  const float  *b0, *bd0, *bd1, *bm, *bu0, *bu1;
  const float  *wo, *bo;
};

inline void run_unet(const UnetW& u, const float* inF32, float* outF32,
                     __bf16* arena1, __bf16* arena2, __bf16* arena3,
                     __bf16* skip0, __bf16* skip1, hipStream_t stream) {
  __bf16* x0 = arena1;   // [4,256,256,128]
  __bf16* h0 = skip0;    // [4,256,256,256]
  __bf16* p0 = arena2;   // [4,128,128,256]
  __bf16* h1 = skip1;    // [4,128,128,512]
  __bf16* p1 = arena3;   // [4,64,64,512]
  __bf16* m  = arena3 + (size_t)4 * 512 * 64 * 64;
  __bf16* u0 = arena2;   // [4,128,128,256]
  __bf16* u1 = arena1;   // [4,256,256,128]
  dim3 blk(TPB);

  k_conv3x3_c1<<<(4 * 128 * 65536) / TPB, blk, 0, stream>>>(
      inF32, u.w0f, u.b0, x0, 4 * 128 * 65536);
  k_conv3x3_wmma<<<dim3(65536 / 64, 256 / 64, 4), blk, 0, stream>>>(
      x0, nullptr, u.wd0, u.bd0, h0, 7, 256, 256, 8, 1, 0);
  k_maxpool2<<<(4 * 128 * 128 * 256 / 8 + TPB - 1) / TPB, blk, 0, stream>>>(
      h0, p0, 8, 8, 8, 4 * 128 * 128 * 256 / 8);
  k_conv3x3_wmma<<<dim3(16384 / 64, 512 / 64, 4), blk, 0, stream>>>(
      p0, nullptr, u.wd1, u.bd1, h1, 8, 512, 128, 7, 1, 0);
  k_maxpool2<<<(4 * 64 * 64 * 512 / 8 + TPB - 1) / TPB, blk, 0, stream>>>(
      h1, p1, 7, 7, 9, 4 * 64 * 64 * 512 / 8);
  k_conv3x3_wmma<<<dim3(4096 / 64, 512 / 64, 4), blk, 0, stream>>>(
      p1, nullptr, u.wm, u.bm, m, 9, 512, 64, 6, 1, 0);
  k_conv3x3_wmma<<<dim3(16384 / 64, 256 / 64, 4), blk, 0, stream>>>(
      h1, m, u.wu0, u.bu0, u0, 9, 256, 128, 7, 1, 1);
  k_conv3x3_wmma<<<dim3(65536 / 64, 128 / 64, 4), blk, 0, stream>>>(
      h0, u0, u.wu1, u.bu1, u1, 8, 128, 256, 8, 1, 1);
  k_conv1x1_f32out<<<(4 * 65536 + TPB - 1) / TPB, blk, 0, stream>>>(
      u1, u.wo, u.bo, outF32, 128, 4 * 65536);
}

} // namespace

extern "C" void kernel_launch(void* const* d_in, const int* in_sizes, int n_in,
                              void* d_out, int out_size, void* d_ws, size_t ws_size,
                              hipStream_t stream) {
  (void)in_sizes; (void)n_in; (void)out_size; (void)ws_size;
  const float* sample = (const float*)d_in[0];
  const float* qc_w   = (const float*)d_in[15];
  const float* qc_b   = (const float*)d_in[16];
  const float* cbk    = (const float*)d_in[17];
  const float* pqc_w  = (const float*)d_in[18];
  const float* pqc_b  = (const float*)d_in[19];

  char* ws = (char*)d_ws;
  size_t off = 0;
  auto alloc = [&](size_t bytes) -> char* {
    char* p = ws + off;
    off += (bytes + 255) & ~(size_t)255;
    return p;
  };

  __bf16* arena1 = (__bf16*)alloc((size_t)134217728);
  __bf16* skip0  = (__bf16*)alloc((size_t)134217728);
  __bf16* skip1  = (__bf16*)alloc((size_t)67108864);
  __bf16* arena2 = (__bf16*)alloc((size_t)67108864);
  __bf16* arena3 = (__bf16*)alloc((size_t)33554432);
  float*  encOut = (float*)alloc((size_t)4 * 65536 * 4);
  float*  decIn  = (float*)alloc((size_t)4 * 65536 * 4);
  float*  lossAc = (float*)alloc(256);

  static const int wsz[5]  = {256 * 128 * 9, 512 * 256 * 9, 512 * 512 * 9,
                              256 * 512 * 9, 128 * 256 * 9};
  static const int wcin[5] = {128, 256, 512, 512, 256};
  static const int wlg[5]  = {7, 8, 9, 9, 8};
  __bf16* wbf[2][5];
  for (int un = 0; un < 2; ++un)
    for (int j = 0; j < 5; ++j)
      wbf[un][j] = (__bf16*)alloc((size_t)wsz[j] * 2);

  dim3 blk(TPB);
  for (int un = 0; un < 2; ++un) {
    const int base = un == 0 ? 1 : 20;
    for (int j = 0; j < 5; ++j) {
      const float* src = (const float*)d_in[base + 2 + 2 * j];
      k_cvt_w_tap<<<(wsz[j] + TPB - 1) / TPB, blk, 0, stream>>>(
          src, wbf[un][j], wcin[j], wlg[j], wsz[j]);
    }
  }

  auto mkU = [&](int un) -> UnetW {
    const int base = un == 0 ? 1 : 20;
    UnetW u;
    u.w0f = (const float*)d_in[base + 0];
    u.wd0 = wbf[un][0]; u.wd1 = wbf[un][1]; u.wm  = wbf[un][2];
    u.wu0 = wbf[un][3]; u.wu1 = wbf[un][4];
    u.b0  = (const float*)d_in[base + 1];
    u.bd0 = (const float*)d_in[base + 3];
    u.bd1 = (const float*)d_in[base + 5];
    u.bm  = (const float*)d_in[base + 7];
    u.bu0 = (const float*)d_in[base + 9];
    u.bu1 = (const float*)d_in[base + 11];
    u.wo  = (const float*)d_in[base + 12];
    u.bo  = (const float*)d_in[base + 13];
    return u;
  };

  run_unet(mkU(0), sample, encOut, arena1, arena2, arena3, skip0, skip1, stream);

  hipMemsetAsync(lossAc, 0, sizeof(float), stream);
  k_vq<<<(4 * 65536) / TPB, blk, 0, stream>>>(encOut, qc_w, qc_b, cbk, pqc_w, pqc_b,
                                              decIn, lossAc);

  run_unet(mkU(1), decIn, (float*)d_out, arena1, arena2, arena3, skip0, skip1, stream);

  k_loss_out<<<1, 1, 0, stream>>>(lossAc, (float*)d_out + 262144);
}
